// Model_88072599372194
// MI455X (gfx1250) — compile-verified, matrix operand order checked by reading, not run on hardware
//
#include <hip/hip_runtime.h>
#include <hip/hip_bf16.h>
#include <math.h>

typedef __attribute__((ext_vector_type(16))) __bf16 v16bf;
typedef __attribute__((ext_vector_type(8)))  float  v8f;

#define B_  4
#define L_  2048
#define D_  512
#define H_  8
#define HD_ 64
#define NL_ 2
#define M_  (B_ * L_)          // 8192 rows
#define EPS_ 1e-5f

#define WMMA_BF16(A, Bv, Cv) \
  __builtin_amdgcn_wmma_f32_16x16x32_bf16(false, (A), false, (Bv), (short)0, (Cv), false, false)

// ---------------------------------------------------------------------------
// LayerNorm over D=512. One wave per row, 16 elems per lane, shfl reductions.
// ---------------------------------------------------------------------------
template <typename OUT>
__global__ __launch_bounds__(256) void ln_kernel(const float* __restrict__ x,
                                                 const float* __restrict__ w,
                                                 const float* __restrict__ bias,
                                                 OUT* __restrict__ out, int rows) {
  int wid  = threadIdx.x >> 5;
  int lane = threadIdx.x & 31;
  int row  = blockIdx.x * 8 + wid;
  if (row >= rows) return;
  const float* xr = x + (size_t)row * D_;
  float v[16];
  float s = 0.f;
#pragma unroll
  for (int i = 0; i < 16; ++i) { v[i] = xr[i * 32 + lane]; s += v[i]; }
#pragma unroll
  for (int m = 16; m >= 1; m >>= 1) s += __shfl_xor(s, m, 32);
  float mu = s * (1.0f / D_);
  float vs = 0.f;
#pragma unroll
  for (int i = 0; i < 16; ++i) { float d = v[i] - mu; vs += d * d; }
#pragma unroll
  for (int m = 16; m >= 1; m >>= 1) vs += __shfl_xor(vs, m, 32);
  float rstd = rsqrtf(vs * (1.0f / D_) + EPS_);
  OUT* orow = out + (size_t)row * D_;
#pragma unroll
  for (int i = 0; i < 16; ++i) {
    int c = i * 32 + lane;
    orow[c] = (OUT)((v[i] - mu) * rstd * w[c] + bias[c]);
  }
}

// ---------------------------------------------------------------------------
// Transpose-cast f32 [K][N] -> bf16 [N][K] (weights, done once per launch)
// ---------------------------------------------------------------------------
__global__ __launch_bounds__(256) void tcast_kernel(const float* __restrict__ src,
                                                    __bf16* __restrict__ dst, int K, int N) {
  int idx = blockIdx.x * blockDim.x + threadIdx.x;
  if (idx >= K * N) return;
  int kk = idx / N, n = idx % N;
  dst[(size_t)n * K + kk] = (__bf16)src[idx];
}

// ---------------------------------------------------------------------------
// GEMM: C[M][N](f32, +opt residual) = A[M][K](bf16 row-major) * Wt[N][K](bf16)
// Block = 256 thr (8 waves). Block tile 128x128, wave tile 32x64 (2x4 accs).
// All fragment addresses are stepped pointers: one v_add per pointer per
// K-step; row deltas fold into instruction immediate offsets. Prefetch is
// unconditional (speculative, safe past the end).
// ---------------------------------------------------------------------------
__global__ __launch_bounds__(256) void gemm_bf16_kernel(const __bf16* __restrict__ A,
                                                        const __bf16* __restrict__ Wt,
                                                        const float* __restrict__ res,
                                                        float* __restrict__ C,
                                                        int N, int K) {
  int wid = threadIdx.x >> 5, lane = threadIdx.x & 31;
  int lh = lane >> 4, lm = lane & 15;
  int m0 = blockIdx.y * 128 + (wid & 3) * 32;
  int n0 = blockIdx.x * 128 + (wid >> 2) * 64;

  const __bf16* ap = A  + (size_t)(m0 + lm) * K + lh * 16;   // rows m0+lm, m0+16+lm
  const __bf16* bp = Wt + (size_t)(n0 + lm) * K + lh * 16;   // rows n0+lm (+16/32/48)
  const size_t rA = (size_t)16 * K;                          // 16-row delta (elems)

  v8f acc[2][4] = {};
  for (int k0 = 0; k0 < K; k0 += 32) {
    v16bf a0 = *(const v16bf*)(ap);
    v16bf a1 = *(const v16bf*)(ap + rA);
    v16bf b0 = *(const v16bf*)(bp);
    v16bf b1 = *(const v16bf*)(bp + rA);
    v16bf b2 = *(const v16bf*)(bp + 2 * rA);
    v16bf b3 = *(const v16bf*)(bp + 3 * rA);
    __builtin_prefetch(ap + 64, 0, 1);
    __builtin_prefetch(bp + 64, 0, 1);
    acc[0][0] = WMMA_BF16(a0, b0, acc[0][0]);
    acc[0][1] = WMMA_BF16(a0, b1, acc[0][1]);
    acc[0][2] = WMMA_BF16(a0, b2, acc[0][2]);
    acc[0][3] = WMMA_BF16(a0, b3, acc[0][3]);
    acc[1][0] = WMMA_BF16(a1, b0, acc[1][0]);
    acc[1][1] = WMMA_BF16(a1, b1, acc[1][1]);
    acc[1][2] = WMMA_BF16(a1, b2, acc[1][2]);
    acc[1][3] = WMMA_BF16(a1, b3, acc[1][3]);
    ap += 32;
    bp += 32;
  }
#pragma unroll
  for (int i = 0; i < 2; ++i)
#pragma unroll
    for (int j = 0; j < 4; ++j)
#pragma unroll
      for (int r = 0; r < 8; ++r) {
        int row = m0 + i * 16 + r + 8 * lh;     // C layout: M = r + 8*(lane/16)
        int col = n0 + j * 16 + lm;             //           N = lane%16
        size_t idx = (size_t)row * N + col;
        float vv = acc[i][j][r];
        if (res) vv += res[idx];
        C[idx] = vv;
      }
}

// ---------------------------------------------------------------------------
// RoPE + layout pack: q,k f32 (B,L,D) -> bf16 (B,H,L,HD); v -> bf16 (B,H,HD,L)
// ---------------------------------------------------------------------------
__global__ __launch_bounds__(256) void rope_pack_kernel(
    const float* __restrict__ q32, const float* __restrict__ k32,
    const float* __restrict__ v32, const float* __restrict__ pcos,
    const float* __restrict__ psin, __bf16* __restrict__ qb,
    __bf16* __restrict__ kb, __bf16* __restrict__ vTb) {
  int idx = blockIdx.x * blockDim.x + threadIdx.x;   // over B*L*D
  if (idx >= M_ * D_) return;
  int c = idx % D_;
  int bl = idx / D_;
  int l = bl % L_, b = bl / L_;
  int h = c >> 6, d = c & 63;
  float cv = pcos[idx], sv = psin[idx];
  float qv = q32[idx], kv = k32[idx];
  float qr = (d < 32) ? -q32[idx + 32] : q32[idx - 32];
  float kr = (d < 32) ? -k32[idx + 32] : k32[idx - 32];
  size_t bh = (size_t)b * H_ + h;
  qb[(bh * L_ + l) * HD_ + d] = (__bf16)(qv * cv + qr * sv);
  kb[(bh * L_ + l) * HD_ + d] = (__bf16)(kv * cv + kr * sv);
  vTb[(bh * HD_ + d) * L_ + l] = (__bf16)v32[idx];
}

// ---------------------------------------------------------------------------
// Flash attention. Grid (L/128, B*H), block 256 (8 waves). Each wave owns 16
// queries; streams keys 32 at a time with stepped pointers. Scores: 4 WMMAs,
// online softmax in f32 (shfl reductions stay in 16-lane halves, matching
// row = r + 8*(lane/16)). P goes C-layout -> A-layout through a wave-private
// LDS slab, then 4 WMMAs accumulate P*V (V pre-transposed to (b,h,d,l)).
// ---------------------------------------------------------------------------
__global__ __launch_bounds__(256) void attn_kernel(const __bf16* __restrict__ q,
                                                   const __bf16* __restrict__ k,
                                                   const __bf16* __restrict__ vT,
                                                   const float* __restrict__ mask,
                                                   __bf16* __restrict__ ctx) {
  __shared__ __align__(64) __bf16 pbuf[8][16][32];
  int wid = threadIdx.x >> 5, lane = threadIdx.x & 31;
  int lh = lane >> 4, lm = lane & 15;
  int bh = blockIdx.y;
  int b = bh / H_, h = bh % H_;
  int q0 = blockIdx.x * 128 + wid * 16;

  const __bf16* qp = q + ((size_t)bh * L_ + q0 + lm) * HD_ + lh * 16;
  v16bf qf0 = *(const v16bf*)(qp);
  v16bf qf1 = *(const v16bf*)(qp + 32);

  // stepped pointers
  const __bf16* kp = k + ((size_t)bh * L_ + lm) * HD_ + lh * 16;  // +32 elem for s=1,
                                                                  // +16*HD_ for j=1
  const __bf16* vp = vT + ((size_t)bh * HD_ + lm) * L_ + lh * 16; // +16*L_ per d-tile
  const float*  mp = mask + (size_t)b * L_ + lm;

  float mrow[8], lrow[8];
#pragma unroll
  for (int r = 0; r < 8; ++r) { mrow[r] = -1e30f; lrow[r] = 0.f; }
  v8f o[4] = {};
  const float scale = 0.125f;  // 1/sqrt(64)

  for (int kt = 0; kt < L_; kt += 32) {
    v8f sa[2] = {};
    {
      v16bf kf00 = *(const v16bf*)(kp);
      v16bf kf01 = *(const v16bf*)(kp + 32);
      v16bf kf10 = *(const v16bf*)(kp + 16 * HD_);
      v16bf kf11 = *(const v16bf*)(kp + 16 * HD_ + 32);
      sa[0] = WMMA_BF16(qf0, kf00, sa[0]);
      sa[0] = WMMA_BF16(qf1, kf01, sa[0]);
      sa[1] = WMMA_BF16(qf0, kf10, sa[1]);
      sa[1] = WMMA_BF16(qf1, kf11, sa[1]);
    }
    float mk0 = mp[0];
    float mk1 = mp[16];
    float corr[8];
#pragma unroll
    for (int r = 0; r < 8; ++r) {
      sa[0][r] = sa[0][r] * scale + mk0;
      sa[1][r] = sa[1][r] * scale + mk1;
      float mx = fmaxf(sa[0][r], sa[1][r]);
#pragma unroll
      for (int m = 8; m >= 1; m >>= 1) mx = fmaxf(mx, __shfl_xor(mx, m, 32));
      float mnew = fmaxf(mrow[r], mx);
      float p0 = __expf(sa[0][r] - mnew);
      float p1 = __expf(sa[1][r] - mnew);
      sa[0][r] = p0; sa[1][r] = p1;
      float ps = p0 + p1;
#pragma unroll
      for (int m = 8; m >= 1; m >>= 1) ps += __shfl_xor(ps, m, 32);
      corr[r] = __expf(mrow[r] - mnew);
      lrow[r] = lrow[r] * corr[r] + ps;
      mrow[r] = mnew;
    }
#pragma unroll
    for (int j = 0; j < 4; ++j)
#pragma unroll
      for (int r = 0; r < 8; ++r) o[j][r] *= corr[r];
    // C-layout -> A-layout for P via wave-private LDS slab
#pragma unroll
    for (int r = 0; r < 8; ++r) {
      pbuf[wid][r + 8 * lh][lm]      = (__bf16)sa[0][r];
      pbuf[wid][r + 8 * lh][16 + lm] = (__bf16)sa[1][r];
    }
    v16bf pa = *(const v16bf*)(&pbuf[wid][lm][lh * 16]);
    {
      v16bf vf0 = *(const v16bf*)(vp);
      v16bf vf1 = *(const v16bf*)(vp + 16 * L_);
      v16bf vf2 = *(const v16bf*)(vp + 32 * L_);
      v16bf vf3 = *(const v16bf*)(vp + 48 * L_);
      o[0] = WMMA_BF16(pa, vf0, o[0]);
      o[1] = WMMA_BF16(pa, vf1, o[1]);
      o[2] = WMMA_BF16(pa, vf2, o[2]);
      o[3] = WMMA_BF16(pa, vf3, o[3]);
    }
    kp += 32 * HD_;
    vp += 32;
    mp += 32;
  }
#pragma unroll
  for (int r = 0; r < 8; ++r) {
    float inv = 1.0f / lrow[r];
    int row = q0 + r + 8 * lh;
#pragma unroll
    for (int j = 0; j < 4; ++j) {
      int col = h * HD_ + j * 16 + lm;
      ctx[((size_t)b * L_ + row) * D_ + col] = (__bf16)(o[j][r] * inv);
    }
  }
}

// ---------------------------------------------------------------------------
// SwiGLU gate: g = silu(h[:, :D]) * h[:, D:2D]  -> bf16
// ---------------------------------------------------------------------------
__global__ __launch_bounds__(256) void swiglu_kernel(const float* __restrict__ hbuf,
                                                     __bf16* __restrict__ g) {
  int idx = blockIdx.x * blockDim.x + threadIdx.x;   // over M*D
  if (idx >= M_ * D_) return;
  int m = idx / D_, j = idx % D_;
  float x1 = hbuf[(size_t)m * (2 * D_) + j];
  float x2 = hbuf[(size_t)m * (2 * D_) + D_ + j];
  g[idx] = (__bf16)(x1 / (1.f + __expf(-x1)) * x2);
}

// ---------------------------------------------------------------------------
extern "C" void kernel_launch(void* const* d_in, const int* in_sizes, int n_in,
                              void* d_out, int out_size, void* d_ws, size_t ws_size,
                              hipStream_t stream) {
  (void)in_sizes; (void)n_in; (void)out_size; (void)ws_size;
  const float* in_x  = (const float*)d_in[0];
  const float* pcos  = (const float*)d_in[1];
  const float* psin  = (const float*)d_in[2];
  const float* mask  = (const float*)d_in[3];
  const float* Wq    = (const float*)d_in[4];
  const float* Wk    = (const float*)d_in[5];
  const float* Wv    = (const float*)d_in[6];
  const float* Wo    = (const float*)d_in[7];
  const float* W1    = (const float*)d_in[8];
  const float* W2    = (const float*)d_in[9];
  const float* ln1_w = (const float*)d_in[10];
  const float* ln1_b = (const float*)d_in[11];
  const float* ln2_w = (const float*)d_in[12];
  const float* ln2_b = (const float*)d_in[13];
  const float* lnf_w = (const float*)d_in[14];
  const float* lnf_b = (const float*)d_in[15];

  char* ws = (char*)d_ws;
  size_t off = 0;
  auto alloc = [&](size_t bytes) -> void* {
    void* p = ws + off;
    off += (bytes + 255) & ~(size_t)255;
    return p;
  };
  float* x   = (float*)alloc((size_t)M_ * D_ * 4);        // running residual
  float* S   = (float*)alloc((size_t)M_ * 2 * D_ * 4);    // q32|k32, reused as h
  float* q32 = S;
  float* k32 = S + (size_t)M_ * D_;
  float* h32 = S;
  float* v32 = (float*)alloc((size_t)M_ * D_ * 4);
  __bf16* xn  = (__bf16*)alloc((size_t)M_ * D_ * 2);      // LN out, reused as ctx
  __bf16* qb  = (__bf16*)alloc((size_t)M_ * D_ * 2);      // reused as gate
  __bf16* kb  = (__bf16*)alloc((size_t)M_ * D_ * 2);
  __bf16* vTb = (__bf16*)alloc((size_t)M_ * D_ * 2);
  __bf16 *WqT[NL_], *WkT[NL_], *WvT[NL_], *WoT[NL_], *W1T[NL_], *W2T[NL_];
  for (int i = 0; i < NL_; ++i) {
    WqT[i] = (__bf16*)alloc((size_t)D_ * D_ * 2);
    WkT[i] = (__bf16*)alloc((size_t)D_ * D_ * 2);
    WvT[i] = (__bf16*)alloc((size_t)D_ * D_ * 2);
    WoT[i] = (__bf16*)alloc((size_t)D_ * D_ * 2);
    W1T[i] = (__bf16*)alloc((size_t)D_ * 2 * D_ * 2);
    W2T[i] = (__bf16*)alloc((size_t)D_ * D_ * 2);
  }

  hipMemcpyAsync(x, in_x, (size_t)M_ * D_ * 4, hipMemcpyDeviceToDevice, stream);

  const int TC = 256;
  int gDD  = (D_ * D_) / TC;        // 1024
  int gD2D = (D_ * 2 * D_) / TC;    // 2048
  for (int i = 0; i < NL_; ++i) {
    tcast_kernel<<<gDD,  TC, 0, stream>>>(Wq + (size_t)i * D_ * D_,     WqT[i], D_, D_);
    tcast_kernel<<<gDD,  TC, 0, stream>>>(Wk + (size_t)i * D_ * D_,     WkT[i], D_, D_);
    tcast_kernel<<<gDD,  TC, 0, stream>>>(Wv + (size_t)i * D_ * D_,     WvT[i], D_, D_);
    tcast_kernel<<<gDD,  TC, 0, stream>>>(Wo + (size_t)i * D_ * D_,     WoT[i], D_, D_);
    tcast_kernel<<<gD2D, TC, 0, stream>>>(W1 + (size_t)i * D_ * 2 * D_, W1T[i], D_, 2 * D_);
    tcast_kernel<<<gDD,  TC, 0, stream>>>(W2 + (size_t)i * D_ * D_,     W2T[i], D_, D_);
  }

  dim3 blk(256);
  dim3 gN512(512 / 128, M_ / 128);    // (4, 64)
  dim3 gN1024(1024 / 128, M_ / 128);  // (8, 64)
  dim3 gAttn(L_ / 128, B_ * H_);      // (16, 32)
  int gElem = (M_ * D_) / TC;         // 16384

  for (int i = 0; i < NL_; ++i) {
    ln_kernel<__bf16><<<M_ / 8, blk, 0, stream>>>(x, ln1_w + i * D_, ln1_b + i * D_, xn, M_);
    gemm_bf16_kernel<<<gN512, blk, 0, stream>>>(xn, WqT[i], nullptr, q32, 512, 512);
    gemm_bf16_kernel<<<gN512, blk, 0, stream>>>(xn, WkT[i], nullptr, k32, 512, 512);
    gemm_bf16_kernel<<<gN512, blk, 0, stream>>>(xn, WvT[i], nullptr, v32, 512, 512);
    rope_pack_kernel<<<gElem, blk, 0, stream>>>(q32, k32, v32, pcos, psin, qb, kb, vTb);
    attn_kernel<<<gAttn, blk, 0, stream>>>(qb, kb, vTb, mask, xn /*ctx*/);
    gemm_bf16_kernel<<<gN512, blk, 0, stream>>>(xn, WoT[i], x, x, 512, 512);
    ln_kernel<__bf16><<<M_ / 8, blk, 0, stream>>>(x, ln2_w + i * D_, ln2_b + i * D_, xn, M_);
    gemm_bf16_kernel<<<gN1024, blk, 0, stream>>>(xn, W1T[i], nullptr, h32, 1024, 512);
    swiglu_kernel<<<gElem, blk, 0, stream>>>(h32, qb /*gate*/);
    gemm_bf16_kernel<<<gN512, blk, 0, stream>>>(qb, W2T[i], x, x, 512, 512);
  }
  ln_kernel<float><<<M_ / 8, blk, 0, stream>>>(x, lnf_w, lnf_b, (float*)d_out, M_);
}
